// RGCNLayer_9182640079550
// MI455X (gfx1250) — matrix operand order repaired and checked by reading, not compiled, not run on hardware
//
#include <hip/hip_runtime.h>

typedef __attribute__((ext_vector_type(16))) _Float16 v16h;
typedef __attribute__((ext_vector_type(8)))  float    v8f;
typedef __attribute__((ext_vector_type(4)))  _Float16 v4h;

#define N_NODES   20000
#define N_EDGES   640000
#define IN_FEAT   128
#define OUT_FEAT  128
#define NUM_RELS  20
#define NUM_BASES 8

// ---------------- kernel 0: h (f32) -> h16 (f16) ----------------
__global__ void k_h2half(const float* __restrict__ h, _Float16* __restrict__ h16, int n) {
  int i = blockIdx.x * blockDim.x + threadIdx.x;
  if (i < n) h16[i] = (_Float16)h[i];
}

// ---------------- kernel 1: basis combination (torch raw-view semantics) ------
// w_rel[r][i][o] = tmp_flat[r*16384 + i*128 + o] where
// tmp[i2][r2][o2] = sum_b w_comp[r2][b] * weight_flat[i2*1024 + b*128 + o2].
// Stored TRANSPOSED as wT16[r][o][i] so the WMMA B-fragment reads contiguous K-pairs.
__global__ void k_basis(const float* __restrict__ weight, const float* __restrict__ w_comp,
                        _Float16* __restrict__ wT16) {
  int f = blockIdx.x * blockDim.x + threadIdx.x;          // [0, R*128*128)
  if (f >= NUM_RELS * IN_FEAT * OUT_FEAT) return;
  int r  = f >> 14;                                       // f / 16384
  int i  = (f >> 7) & 127;
  int o  = f & 127;
  int i2 = f / (NUM_RELS * OUT_FEAT);                     // f / 2560
  int r2 = (f >> 7) % NUM_RELS;
  float acc = 0.f;
#pragma unroll
  for (int b = 0; b < NUM_BASES; ++b)
    acc += w_comp[r2 * NUM_BASES + b] * weight[i2 * (NUM_BASES * OUT_FEAT) + b * OUT_FEAT + o];
  wT16[(size_t)r * (IN_FEAT * OUT_FEAT) + (size_t)o * IN_FEAT + i] = (_Float16)acc;
}

// ---------------- kernel 2: xW16[n][r][o] = h16[n][:] @ w_rel[r]  (WMMA f16) ---
// Block = 256 threads (8 waves). Block handles one 16-node tile for one relation;
// wave w computes output columns [16w, 16w+16). 4 K-chunks of 32.
__global__ void __launch_bounds__(256)
k_gemm(const _Float16* __restrict__ h16, const _Float16* __restrict__ wT16,
       _Float16* __restrict__ xW16) {
  const int wave = threadIdx.x >> 5;
  const int lane = threadIdx.x & 31;
  const int g    = lane >> 4;     // half-wave group
  const int li   = lane & 15;     // M (for A) / N (for B) within tile
  const int t    = blockIdx.x;    // node tile
  const int r    = blockIdx.y;    // relation

  const _Float16* aBase = h16  + (size_t)(t * 16 + li) * IN_FEAT;
  const _Float16* bBase = wT16 + (size_t)r * (IN_FEAT * OUT_FEAT)
                               + (size_t)(wave * 16 + li) * IN_FEAT;

  v8f c = {0.f, 0.f, 0.f, 0.f, 0.f, 0.f, 0.f, 0.f};

#pragma unroll
  for (int kc = 0; kc < IN_FEAT; kc += 32) {
    union { unsigned u[8]; v16h v; } a, b;
#pragma unroll
    for (int q = 0; q < 8; ++q) {
      // A 16x32 f16 layout: VGPR q -> K = (q>=4?16:0) + g*8 + 2*(q&3)
      int ka = kc + ((q & 4) << 2) + g * 8 + ((q & 3) << 1);
      a.u[q] = *(const unsigned*)(aBase + ka);
      // B 32x16 f16 layout: VGPR q -> K = (q>=4?16:0) + g*8 + 2*(q&3) (column = li)
      int kb = kc + ((q & 4) << 2) + g * 8 + ((q & 3) << 1);
      b.u[q] = *(const unsigned*)(bBase + kb);
    }
    c = __builtin_amdgcn_wmma_f32_16x16x32_f16(false, a.v, false, b.v,
                                               (short)0, c, false, false);
  }

  // D layout: VGPR p, lanes 0-15 -> M=p, lanes 16-31 -> M=p+8; N = li.
  _Float16* oBase = xW16 + (size_t)(t * 16 + g * 8) * (NUM_RELS * OUT_FEAT)
                         + r * OUT_FEAT + wave * 16 + li;
#pragma unroll
  for (int p = 0; p < 8; ++p)
    oBase[(size_t)p * (NUM_RELS * OUT_FEAT)] = (_Float16)c[p];
}

// ---------------- kernel 3: zero output ----------------
__global__ void k_zero(float4* __restrict__ out, int n4) {
  int i = blockIdx.x * blockDim.x + threadIdx.x;
  if (i < n4) out[i] = make_float4(0.f, 0.f, 0.f, 0.f);
}

// ---------------- kernel 4: edge gather + atomic scatter-sum ----------------
// One wave per edge: lane loads 4 f16 msg values (8B), does 4 f32 atomic adds.
__global__ void k_scatter(const _Float16* __restrict__ xW16, const int* __restrict__ src,
                          const int* __restrict__ dst, const int* __restrict__ ety,
                          float* __restrict__ out) {
  int gid  = blockIdx.x * blockDim.x + threadIdx.x;
  int e    = gid >> 5;
  int lane = gid & 31;
  if (e >= N_EDGES) return;
  int s = src[e], d = dst[e], t = ety[e];
  const v4h m = *(const v4h*)(xW16 + (size_t)s * (NUM_RELS * OUT_FEAT)
                                   + t * OUT_FEAT + lane * 4);
  float* o = out + (size_t)d * OUT_FEAT + lane * 4;
  unsafeAtomicAdd(o + 0, (float)m.x);
  unsafeAtomicAdd(o + 1, (float)m.y);
  unsafeAtomicAdd(o + 2, (float)m.z);
  unsafeAtomicAdd(o + 3, (float)m.w);
}

// ---------------- kernel 5: in-place ReLU ----------------
__global__ void k_relu(float* __restrict__ out, int n) {
  int i = blockIdx.x * blockDim.x + threadIdx.x;
  if (i < n) out[i] = fmaxf(out[i], 0.f);
}

extern "C" void kernel_launch(void* const* d_in, const int* in_sizes, int n_in,
                              void* d_out, int out_size, void* d_ws, size_t ws_size,
                              hipStream_t stream) {
  const float* h      = (const float*)d_in[0];   // [20000,128]
  const float* weight = (const float*)d_in[1];   // [8,128,128]
  const float* w_comp = (const float*)d_in[2];   // [20,8]
  const int*   src    = (const int*)d_in[3];     // [640000]
  const int*   dst    = (const int*)d_in[4];
  const int*   ety    = (const int*)d_in[5];
  float*       out    = (float*)d_out;           // [20000,128]

  // Workspace layout (4KB-aligned offsets):
  //   h16  : 20000*128*2      = 5,120,000 B @ 0
  //   wT16 : 20*128*128*2     =   655,360 B @ 5,120,000
  //   xW16 : 20000*20*128*2   = 102,400,000 B @ 5,775,360
  char* ws = (char*)d_ws;
  _Float16* h16  = (_Float16*)(ws);
  _Float16* wT16 = (_Float16*)(ws + 5120000);
  _Float16* xW16 = (_Float16*)(ws + 5775360);

  const int nH = N_NODES * IN_FEAT;                   // 2,560,000
  k_h2half<<<(nH + 255) / 256, 256, 0, stream>>>(h, h16, nH);

  const int nW = NUM_RELS * IN_FEAT * OUT_FEAT;       // 327,680
  k_basis<<<(nW + 255) / 256, 256, 0, stream>>>(weight, w_comp, wT16);

  dim3 gGemm(N_NODES / 16, NUM_RELS);                 // (1250, 20)
  k_gemm<<<gGemm, 256, 0, stream>>>(h16, wT16, xW16);

  const int nOut  = N_NODES * OUT_FEAT;               // 2,560,000
  const int nOut4 = nOut / 4;
  k_zero<<<(nOut4 + 255) / 256, 256, 0, stream>>>((float4*)out, nOut4);

  const long long nThr = (long long)N_EDGES * 32;     // one wave per edge
  k_scatter<<<(unsigned)((nThr + 255) / 256), 256, 0, stream>>>(xW16, src, dst, ety, out);

  k_relu<<<(nOut + 255) / 256, 256, 0, stream>>>(out, nOut);
}
